// MultiHeadAttention_68040871903562
// MI455X (gfx1250) — matrix operand order repaired and checked by reading, not compiled, not run on hardware
//
#include <hip/hip_runtime.h>
#include <hip/hip_bf16.h>

// Problem constants (from reference)
#define B_  4
#define T_  2048
#define D_  1024
#define H_  16
#define HD_ 64
#define M_  (B_ * T_)   // 8192 rows

typedef __attribute__((ext_vector_type(16))) __bf16      v16bf;
typedef __attribute__((ext_vector_type(8)))  float       v8f;
typedef __attribute__((ext_vector_type(8)))  unsigned int v8u;

__device__ __forceinline__ unsigned short f32_bf16(float f) {
  unsigned int u = __builtin_bit_cast(unsigned int, f);
  u += 0x7FFFu + ((u >> 16) & 1u);           // round-to-nearest-even
  return (unsigned short)(u >> 16);
}

// A-fragment (16x32 bf16) per-lane K offsets for vector element pair 2*j
// (contiguous order: k = 16*half + 2*j, matching the B fragments)
__device__ __forceinline__ int a_off(int j, int half) {
  return half * 16 + 2 * j;
}

__device__ __forceinline__ unsigned int ld_u32(const unsigned short* p) {
  unsigned int v;
  __builtin_memcpy(&v, p, 4);
  return v;
}

// ---------------- conversion kernels ----------------
__global__ void cvt_f32_bf16_v4(const float* __restrict__ src,
                                unsigned short* __restrict__ dst, int n) {
  int i = (blockIdx.x * blockDim.x + threadIdx.x) * 4;
  if (i < n) {
    float4 f;
    __builtin_memcpy(&f, src + i, 16);
    unsigned long long p =
        (unsigned long long)f32_bf16(f.x) |
        ((unsigned long long)f32_bf16(f.y) << 16) |
        ((unsigned long long)f32_bf16(f.z) << 32) |
        ((unsigned long long)f32_bf16(f.w) << 48);
    __builtin_memcpy(dst + i, &p, 8);
  }
}

// src [rows=K, cols=N] f32  ->  dst [N, K] bf16 (transposed)
__global__ void cvt_transpose_f32_bf16(const float* __restrict__ src,
                                       unsigned short* __restrict__ dst,
                                       int rows, int cols) {
  int i = blockIdx.x * blockDim.x + threadIdx.x;
  if (i < rows * cols) {
    int k = i / cols;
    int c = i % cols;
    dst[(size_t)c * rows + k] = f32_bf16(src[i]);
  }
}

// ---------------- WMMA GEMM with async-to-LDS B staging ----------------
// A:[M,K] bf16 row-major, Bt:[N,K] bf16 (B transposed) -> out [M,N]
// Block = 128x64 output tile (8 waves x 16x64, shared B columns).
// B k-tile (64 n x 32 k = 4KB) staged via global_load_async_to_lds_b128,
// double buffered with s_wait_asynccnt.
// MODE 0: bf16 plain   MODE 1: bf16, V-transposed store [B,H,HD,T]
// MODE 2: f32 + bias
template <int MODE>
__global__ __launch_bounds__(256) void gemm_wmma(
    const unsigned short* __restrict__ A,
    const unsigned short* __restrict__ Bt,
    void* __restrict__ outp,
    const float* __restrict__ bias,
    int Mdim, int Ndim, int Kdim) {
  __shared__ __align__(16) unsigned short tileB[2][64 * 32];

  int wave = threadIdx.x >> 5;
  int lane = threadIdx.x & 31;
  int tiles_n = Ndim >> 6;
  int bm = blockIdx.x / tiles_n;
  int tn = blockIdx.x % tiles_n;
  int tm = bm * 8 + wave;
  int half = lane >> 4, ln = lane & 15;

  const unsigned short* arow = A + (size_t)(tm * 16 + ln) * Kdim;

  // async staging: thread t copies 16B; row = t/4, 16B segment = t%4
  int t = threadIdx.x;
  const unsigned short* bsrc =
      Bt + (size_t)(tn * 64 + (t >> 2)) * Kdim + (t & 3) * 8;
  unsigned lds_dst0 = (unsigned)(size_t)(&tileB[0][0]) + (unsigned)t * 16u;
  unsigned lds_dst1 = (unsigned)(size_t)(&tileB[1][0]) + (unsigned)t * 16u;

  // prologue: stage k-chunk 0 into buffer 0
  {
    unsigned long long ga = (unsigned long long)(size_t)bsrc;
    asm volatile("global_load_async_to_lds_b128 %0, %1, off"
                 :: "v"(lds_dst0), "v"(ga) : "memory");
  }

  v8f acc[4];
#pragma unroll
  for (int nt = 0; nt < 4; ++nt)
#pragma unroll
    for (int r = 0; r < 8; ++r) acc[nt][r] = 0.f;

  int niter = Kdim >> 5;
  for (int i = 0; i < niter; ++i) {
    int k0 = i << 5;
    if (i + 1 < niter) {
      unsigned long long ga = (unsigned long long)(size_t)(bsrc + k0 + 32);
      unsigned dst = ((i + 1) & 1) ? lds_dst1 : lds_dst0;
      asm volatile("global_load_async_to_lds_b128 %0, %1, off"
                   :: "v"(dst), "v"(ga) : "memory");
      asm volatile("s_wait_asynccnt 0x1" ::: "memory");
    } else {
      asm volatile("s_wait_asynccnt 0x0" ::: "memory");
    }
    __syncthreads();  // current buffer fully written & visible

    v8u au;
#pragma unroll
    for (int j = 0; j < 8; ++j) au[j] = ld_u32(arow + k0 + a_off(j, half));
    v16bf av = __builtin_bit_cast(v16bf, au);
    if (k0 + 32 < Kdim) __builtin_prefetch(arow + k0 + 32, 0, 0);

    const unsigned short* bl = tileB[i & 1];
#pragma unroll
    for (int nt = 0; nt < 4; ++nt) {
      v8u bu;
#pragma unroll
      for (int j = 0; j < 8; ++j)
        bu[j] = ld_u32(bl + (nt * 16 + ln) * 32 + half * 16 + 2 * j);
      v16bf bv = __builtin_bit_cast(v16bf, bu);
      acc[nt] = __builtin_amdgcn_wmma_f32_16x16x32_bf16(
          false, av, false, bv, (short)0, acc[nt], false, false);
    }
    __syncthreads();  // all waves done reading before buffer reuse
  }

#pragma unroll
  for (int nt = 0; nt < 4; ++nt) {
#pragma unroll
    for (int r = 0; r < 8; ++r) {
      int row = tm * 16 + half * 8 + r;
      int col = tn * 64 + nt * 16 + ln;
      float v = acc[nt][r];
      if (MODE == 0) {
        ((unsigned short*)outp)[(size_t)row * Ndim + col] = f32_bf16(v);
      } else if (MODE == 1) {
        int b = row / T_, tt = row % T_;
        int h = col / HD_, d = col % HD_;
        ((unsigned short*)outp)[(((size_t)(b * H_ + h) * HD_) + d) * T_ + tt] =
            f32_bf16(v);
      } else {
        ((float*)outp)[(size_t)row * Ndim + col] = v + bias[col];
      }
    }
  }
}

// ---------------- flash attention ----------------
// Q,K:[B*T,D] bf16; Vt:[B*H,HD,T] bf16; Ctx:[B*T,D] bf16
__global__ __launch_bounds__(256) void attn_flash(
    const unsigned short* __restrict__ Q,
    const unsigned short* __restrict__ Km,
    const unsigned short* __restrict__ Vt,
    unsigned short* __restrict__ Ctx) {
  __shared__ unsigned short plds[8][16 * 32];  // per-wave P tile (bf16)
  int wave = threadIdx.x >> 5, lane = threadIdx.x & 31;
  int tile = blockIdx.x * 8 + wave;
  const int QT = T_ / 16;
  int qt = tile % QT;
  int bh = tile / QT;
  int b = bh / H_, h = bh % H_;
  int q0 = qt * 16;
  int half = lane >> 4, ln = lane & 15;

  // Q fragments: K-dim = HD = 64 -> two 32-wide chunks, loaded once
  const unsigned short* qrow = Q + (size_t)(b * T_ + q0 + ln) * D_ + h * HD_;
  v16bf aq[2];
#pragma unroll
  for (int c = 0; c < 2; ++c) {
    v8u u;
#pragma unroll
    for (int j = 0; j < 8; ++j) u[j] = ld_u32(qrow + 32 * c + a_off(j, half));
    aq[c] = __builtin_bit_cast(v16bf, u);
  }

  v8f acc[4];
#pragma unroll
  for (int dt = 0; dt < 4; ++dt)
#pragma unroll
    for (int r = 0; r < 8; ++r) acc[dt][r] = 0.f;

  float mrow[8], lrow[8];
#pragma unroll
  for (int r = 0; r < 8; ++r) {
    mrow[r] = -__builtin_inff();
    lrow[r] = 0.f;
  }

  // work in log2 domain: exp2f is the native v_exp_f32
  const float sl2e = 0.125f * 1.44269504088896340736f;  // (1/sqrt(64))*log2(e)
  unsigned short* pl = plds[wave];
  const unsigned short* vbase = Vt + (size_t)bh * HD_ * T_;

  int nsteps = (q0 + 15) / 32 + 1;  // causal: cover cols up to q0+15
  for (int s = 0; s < nsteps; ++s) {
    int kbase = s * 32;
    // ---- S = Q K^T for 32 key columns (two 16-col halves) ----
    v8f sc[2];
#pragma unroll
    for (int ch = 0; ch < 2; ++ch) {
#pragma unroll
      for (int r = 0; r < 8; ++r) sc[ch][r] = 0.f;
      const unsigned short* krow =
          Km + (size_t)(b * T_ + kbase + ch * 16 + ln) * D_ + h * HD_;
#pragma unroll
      for (int c = 0; c < 2; ++c) {
        v8u bu;
#pragma unroll
        for (int j = 0; j < 8; ++j)
          bu[j] = ld_u32(krow + 32 * c + half * 16 + 2 * j);
        v16bf bv = __builtin_bit_cast(v16bf, bu);
        sc[ch] = __builtin_amdgcn_wmma_f32_16x16x32_bf16(
            false, aq[c], false, bv, (short)0, sc[ch], false, false);
      }
    }
    // ---- scale + causal mask + online softmax (width-16 row reductions) ----
    float p0s[8], p1s[8];
#pragma unroll
    for (int r = 0; r < 8; ++r) {
      int row = q0 + half * 8 + r;
      float s0 = sc[0][r] * sl2e;
      float s1 = sc[1][r] * sl2e;
      if (kbase + ln > row) s0 = -__builtin_inff();
      if (kbase + 16 + ln > row) s1 = -__builtin_inff();
      float mx = fmaxf(s0, s1);
#pragma unroll
      for (int off = 8; off; off >>= 1) mx = fmaxf(mx, __shfl_xor(mx, off, 16));
      float mnew = fmaxf(mrow[r], mx);
      float corr = exp2f(mrow[r] - mnew);
      float p0 = exp2f(s0 - mnew);
      float p1 = exp2f(s1 - mnew);
      float sum = p0 + p1;
#pragma unroll
      for (int off = 8; off; off >>= 1) sum += __shfl_xor(sum, off, 16);
      lrow[r] = lrow[r] * corr + sum;
      mrow[r] = mnew;
      p0s[r] = p0;
      p1s[r] = p1;
#pragma unroll
      for (int dt = 0; dt < 4; ++dt) acc[dt][r] *= corr;
    }
    // ---- relayout P (C-format) -> A-format via per-wave LDS tile ----
#pragma unroll
    for (int r = 0; r < 8; ++r) {
      pl[(half * 8 + r) * 32 + ln] = f32_bf16(p0s[r]);
      pl[(half * 8 + r) * 32 + 16 + ln] = f32_bf16(p1s[r]);
    }
    v8u pu;
#pragma unroll
    for (int j = 0; j < 8; ++j) {
      unsigned int x;
      __builtin_memcpy(&x, &pl[ln * 32 + a_off(j, half)], 4);
      pu[j] = x;
    }
    v16bf pa = __builtin_bit_cast(v16bf, pu);
    // ---- acc += P @ V (K-dim = 32 keys, 4 d-tiles of 16) ----
#pragma unroll
    for (int dt = 0; dt < 4; ++dt) {
      const unsigned short* vr = vbase + (size_t)(dt * 16 + ln) * T_ + kbase;
      v8u vu;
#pragma unroll
      for (int j = 0; j < 8; ++j) vu[j] = ld_u32(vr + half * 16 + 2 * j);
      v16bf vv = __builtin_bit_cast(v16bf, vu);
      acc[dt] = __builtin_amdgcn_wmma_f32_16x16x32_bf16(
          false, pa, false, vv, (short)0, acc[dt], false, false);
    }
  }
  // ---- normalize + store ctx (bf16) ----
#pragma unroll
  for (int dt = 0; dt < 4; ++dt)
#pragma unroll
    for (int r = 0; r < 8; ++r) {
      int row = q0 + half * 8 + r;
      int col = h * HD_ + dt * 16 + ln;
      Ctx[(size_t)(b * T_ + row) * D_ + col] = f32_bf16(acc[dt][r] / lrow[r]);
    }
}

// ---------------- launch ----------------
extern "C" void kernel_launch(void* const* d_in, const int* in_sizes, int n_in,
                              void* d_out, int out_size, void* d_ws,
                              size_t ws_size, hipStream_t stream) {
  const float* x  = (const float*)d_in[0];
  const float* Wq = (const float*)d_in[1];
  const float* Wk = (const float*)d_in[2];
  const float* Wv = (const float*)d_in[3];
  const float* Wo = (const float*)d_in[4];
  const float* bo = (const float*)d_in[5];
  float* out = (float*)d_out;

  char* ws = (char*)d_ws;
  size_t off = 0;
  unsigned short* xb  = (unsigned short*)(ws + off); off += (size_t)M_ * D_ * 2;
  unsigned short* wqt = (unsigned short*)(ws + off); off += (size_t)D_ * D_ * 2;
  unsigned short* wkt = (unsigned short*)(ws + off); off += (size_t)D_ * D_ * 2;
  unsigned short* wvt = (unsigned short*)(ws + off); off += (size_t)D_ * D_ * 2;
  unsigned short* wot = (unsigned short*)(ws + off); off += (size_t)D_ * D_ * 2;
  unsigned short* qb  = (unsigned short*)(ws + off); off += (size_t)M_ * D_ * 2;
  unsigned short* kb  = (unsigned short*)(ws + off); off += (size_t)M_ * D_ * 2;
  unsigned short* vtb = (unsigned short*)(ws + off); off += (size_t)M_ * D_ * 2;
  unsigned short* cb  = (unsigned short*)(ws + off); off += (size_t)M_ * D_ * 2;

  int n = M_ * D_;
  cvt_f32_bf16_v4<<<(n / 4 + 255) / 256, 256, 0, stream>>>(x, xb, n);
  int nw = D_ * D_;
  cvt_transpose_f32_bf16<<<(nw + 255) / 256, 256, 0, stream>>>(Wq, wqt, D_, D_);
  cvt_transpose_f32_bf16<<<(nw + 255) / 256, 256, 0, stream>>>(Wk, wkt, D_, D_);
  cvt_transpose_f32_bf16<<<(nw + 255) / 256, 256, 0, stream>>>(Wv, wvt, D_, D_);
  cvt_transpose_f32_bf16<<<(nw + 255) / 256, 256, 0, stream>>>(Wo, wot, D_, D_);

  int gblocks = (M_ / 128) * (D_ / 64);  // 64 * 16 = 1024 blocks
  gemm_wmma<0><<<gblocks, 256, 0, stream>>>(xb, wqt, qb, nullptr, M_, D_, D_);
  gemm_wmma<0><<<gblocks, 256, 0, stream>>>(xb, wkt, kb, nullptr, M_, D_, D_);
  gemm_wmma<1><<<gblocks, 256, 0, stream>>>(xb, wvt, vtb, nullptr, M_, D_, D_);

  int ablocks = (B_ * H_ * (T_ / 16)) / 8;  // 1024 blocks x 8 waves
  attn_flash<<<ablocks, 256, 0, stream>>>(qb, kb, vtb, cb);

  gemm_wmma<2><<<gblocks, 256, 0, stream>>>(cb, wot, out, bo, M_, D_, D_);
}
